// OptimizedGAT_83193516524086
// MI455X (gfx1250) — matrix operand order repaired and checked by reading, not compiled, not run on hardware
//
#include <hip/hip_runtime.h>
#include <math.h>

typedef __attribute__((ext_vector_type(16))) _Float16 v16h;
typedef __attribute__((ext_vector_type(8)))  _Float16 v8h;
typedef __attribute__((ext_vector_type(8)))  float    v8f;

#define IN_CH 128
#define HID   256
#define HEADS 8
#define CDIM  32
#define NLAYERS 4

// ------------------------------------------------------------------
// Weight prep: W[K x Nc] f32 (row-major) -> Wt[Nc x K] f16 (row-major)
// ------------------------------------------------------------------
__global__ void k_transpose_f16(const float* __restrict__ W, _Float16* __restrict__ Wt,
                                int K, int Nc) {
  int idx = blockIdx.x * blockDim.x + threadIdx.x;
  if (idx >= K * Nc) return;
  int n = idx / K;
  int k = idx - n * K;
  Wt[(size_t)n * K + k] = (_Float16)W[(size_t)k * Nc + n];
}

// Fold bias + eval-BatchNorm into per-column scale/shift:
//   y = (acc + bias)*s + t  with  s = g/sqrt(1+eps), t = bn_b   (or s=1,t=0)
__global__ void k_make_ss(const float* __restrict__ bias, const float* __restrict__ bng,
                          const float* __restrict__ bnb, float* __restrict__ scale,
                          float* __restrict__ shift, int n, int has_bn) {
  int i = blockIdx.x * blockDim.x + threadIdx.x;
  if (i >= n) return;
  float s = 1.0f, t = 0.0f;
  if (has_bn) { s = bng[i] * rsqrtf(1.0f + 1e-5f); t = bnb[i]; }
  scale[i] = s;
  shift[i] = bias[i] * s + t;
}

// ------------------------------------------------------------------
// Generic WMMA GEMM:  C[M x Nc] = epilogue( A[M x K] @ Wt^T )
//   A: f32 row-major, Wt: f16 [Nc x K] (W transposed)
// Block: 256 threads = 8 waves; tile 128(M) x 64(N), K-step 32.
// Each wave: one 16-row strip, four 16-col tiles (4 WMMA accumulators),
// double-buffered LDS (one barrier per K-iteration).
// ------------------------------------------------------------------
#define BM 128
#define BN 64
#define BK 32

__device__ __forceinline__ void stage_tiles(
    const float* __restrict__ A, const _Float16* __restrict__ Wt,
    _Float16 (*sA)[BK], _Float16 (*sB)[BK],
    int tid, int row0, int col0, int M, int K, int k0)
{
  // A tile: 128 rows x 32 k, f32 -> f16, 16 elems/thread
  {
    const int arow = tid >> 1;
    const int akh  = (tid & 1) * 16;
    const int gr   = row0 + arow;
    _Float16* dst = &sA[arow][akh];
    if (gr < M) {
      const float* src = A + (size_t)gr * K + k0 + akh;
      if (k0 + 2 * BK < K) __builtin_prefetch(src + 2 * BK, 0, 3);
      #pragma unroll
      for (int i = 0; i < 16; ++i) dst[i] = (_Float16)src[i];
    } else {
      #pragma unroll
      for (int i = 0; i < 16; ++i) dst[i] = (_Float16)0.0f;
    }
  }
  // B tile: 64 cols x 32 k, f16 copy (contiguous in K)
  if (tid < 128) {
    const int n  = tid >> 1;
    const int kh = (tid & 1) * 16;
    const _Float16* src = Wt + (size_t)(col0 + n) * K + k0 + kh;
    *(v8h*)(&sB[n][kh])     = *(const v8h*)(src);
    *(v8h*)(&sB[n][kh + 8]) = *(const v8h*)(src + 8);
  }
}

__global__ __launch_bounds__(256) void k_gemm_wmma(
    const float* __restrict__ A, const _Float16* __restrict__ Wt,
    float* __restrict__ C, const float* __restrict__ scale,
    const float* __restrict__ shift, int M, int K, int Nc, int act)
{
  __shared__ __align__(16) _Float16 sA[2][BM][BK];
  __shared__ __align__(16) _Float16 sB[2][BN][BK];
  const int tid  = threadIdx.x;
  const int wave = tid >> 5;
  const int lane = tid & 31;
  const int row0 = blockIdx.x * BM;
  const int col0 = blockIdx.y * BN;

  v8f acc[4];
  #pragma unroll
  for (int ct = 0; ct < 4; ++ct) acc[ct] = (v8f){};

  const int nIter = K / BK;

  stage_tiles(A, Wt, sA[0], sB[0], tid, row0, col0, M, K, 0);
  __syncthreads();

  for (int it = 0; it < nIter; ++it) {
    const int p = it & 1;

    // A fragment (16x32 f16): lane -> row m=lane&15;
    // K chunks {kb..kb+7} and {kb+16..kb+23}, kb = 8*(lane>=16)
    const int mrow = wave * 16 + (lane & 15);
    const int kb   = (lane & 16) ? 8 : 0;
    v8h a0 = *(const v8h*)(&sA[p][mrow][kb]);
    v8h a1 = *(const v8h*)(&sA[p][mrow][kb + 16]);
    v16h afrag = __builtin_shufflevector(a0, a1,
        0,1,2,3,4,5,6,7,8,9,10,11,12,13,14,15);

    // B fragments: lane -> col n=lane&15; K chunks {kc..kc+7} and {kc+16..kc+23}, kc = 8*(lane>=16)
    const int nbase = lane & 15;
    const int kc    = (lane & 16) ? 8 : 0;
    v16h bfrag[4];
    #pragma unroll
    for (int ct = 0; ct < 4; ++ct) {
      v8h b0 = *(const v8h*)(&sB[p][ct * 16 + nbase][kc]);
      v8h b1 = *(const v8h*)(&sB[p][ct * 16 + nbase][kc + 16]);
      bfrag[ct] = __builtin_shufflevector(b0, b1,
          0,1,2,3,4,5,6,7,8,9,10,11,12,13,14,15);
    }

    // stage next tile into the other buffer while WMMAs run
    if (it + 1 < nIter)
      stage_tiles(A, Wt, sA[p ^ 1], sB[p ^ 1], tid, row0, col0, M, K, (it + 1) * BK);

    #pragma unroll
    for (int ct = 0; ct < 4; ++ct)
      acc[ct] = __builtin_amdgcn_wmma_f32_16x16x32_f16(false, afrag, false, bfrag[ct],
                                                       (short)0, acc[ct], false, false);
    __syncthreads();
  }

  // ---- epilogue: D layout: n = lane&15, m = r + 8*(lane>=16) ----
  const int nloc = lane & 15;
  const int mhi  = (lane >> 4) * 8;
  #pragma unroll
  for (int ct = 0; ct < 4; ++ct) {
    int col = col0 + ct * 16 + nloc;
    float sc = scale[col];
    float sh = shift[col];
    #pragma unroll
    for (int r = 0; r < 8; ++r) {
      int grow = row0 + wave * 16 + mhi + r;
      if (grow < M) {
        float v = acc[ct][r] * sc + sh;
        if (act) v = (v > 0.0f) ? v : 0.1f * v;
        C[(size_t)grow * Nc + col] = v;
      }
    }
  }
}

// ------------------------------------------------------------------
// GAT edge kernels
// ------------------------------------------------------------------
__device__ __forceinline__ void atomicMaxF(float* addr, float val) {
  // ordering trick: int-max for val>=0, uint-min for val<0 (init = -inf)
  if (val >= 0.0f) atomicMax((int*)addr, __float_as_int(val));
  else             atomicMin((unsigned int*)addr, __float_as_uint(val));
}

__global__ void k_gat_init(float* __restrict__ m, float* __restrict__ s,
                           float* __restrict__ o, int N) {
  int i = blockIdx.x * blockDim.x + threadIdx.x;
  if (i < N * HEADS) { m[i] = -__builtin_inff(); s[i] = 0.0f; }
  if (i < N * HID)   o[i] = 0.0f;
}

// pass 1: logits[e,h] = att_h . lrelu0.2(xl[src] + xr[dst]); atomic max into m[dst,h]
__global__ void k_edge_logits(const int* __restrict__ ei, int E, int N,
                              const float* __restrict__ xl, const float* __restrict__ xr,
                              const float* __restrict__ att,
                              float* __restrict__ logits, float* __restrict__ mbuf) {
  int t = blockIdx.x * blockDim.x + threadIdx.x;
  int E2 = E + N;
  if (t >= E2 * HEADS) return;
  int e = t >> 3, h = t & 7;
  int src, dst;
  if (e < E) { src = ei[e]; dst = ei[E + e]; } else { src = dst = e - E; }
  const float* pl = xl + (size_t)src * HID + h * CDIM;
  const float* pr = xr + (size_t)dst * HID + h * CDIM;
  const float* pa = att + h * CDIM;
  float acc = 0.0f;
  #pragma unroll
  for (int c = 0; c < CDIM; ++c) {
    float v = pl[c] + pr[c];
    v = (v > 0.0f) ? v : 0.2f * v;
    acc += pa[c] * v;
  }
  logits[t] = acc;
  atomicMaxF(&mbuf[dst * HEADS + h], acc);
}

// pass 2: a = exp(logit - m[dst]); store in place; atomicAdd to s[dst,h]
__global__ void k_edge_exp(const int* __restrict__ ei, int E, int N,
                           float* __restrict__ logits, const float* __restrict__ mbuf,
                           float* __restrict__ sbuf) {
  int t = blockIdx.x * blockDim.x + threadIdx.x;
  int E2 = E + N;
  if (t >= E2 * HEADS) return;
  int e = t >> 3, h = t & 7;
  int dst = (e < E) ? ei[E + e] : (e - E);
  float a = expf(logits[t] - mbuf[dst * HEADS + h]);
  logits[t] = a;
  atomicAdd(&sbuf[dst * HEADS + h], a);
}

// pass 3: o[dst,h,:] += xl[src,h,:] * (a / (s[dst,h] + 1e-16))
__global__ void k_edge_aggr(const int* __restrict__ ei, int E, int N,
                            const float* __restrict__ abuf, const float* __restrict__ sbuf,
                            const float* __restrict__ xl, float* __restrict__ o) {
  int t = blockIdx.x * blockDim.x + threadIdx.x;
  int E2 = E + N;
  if (t >= E2 * HEADS) return;
  int e = t >> 3, h = t & 7;
  int src, dst;
  if (e < E) { src = ei[e]; dst = ei[E + e]; } else { src = dst = e - E; }
  float alpha = abuf[t] / (sbuf[dst * HEADS + h] + 1e-16f);
  const float* pl = xl + (size_t)src * HID + h * CDIM;
  float* po = o + (size_t)dst * HID + h * CDIM;
  #pragma unroll
  for (int c = 0; c < CDIM; ++c) atomicAdd(&po[c], pl[c] * alpha);
}

// g[n,:] += lrelu0.1( LayerNorm(o[n,:] + gbias) )   (one node per 256-thread block)
__global__ __launch_bounds__(256) void k_ln_res(
    const float* __restrict__ o, const float* __restrict__ gbias,
    const float* __restrict__ lng, const float* __restrict__ lnb,
    float* __restrict__ g) {
  __shared__ float red[8];
  int n = blockIdx.x;
  int c = threadIdx.x;
  size_t base = (size_t)n * HID;
  float v = o[base + c] + gbias[c];
  float s1 = v;
  #pragma unroll
  for (int off = 16; off > 0; off >>= 1) s1 += __shfl_xor(s1, off, 32);
  if ((c & 31) == 0) red[c >> 5] = s1;
  __syncthreads();
  float mu = 0.0f;
  #pragma unroll
  for (int i = 0; i < 8; ++i) mu += red[i];
  mu *= (1.0f / (float)HID);
  __syncthreads();
  float d = v - mu;
  float s2 = d * d;
  #pragma unroll
  for (int off = 16; off > 0; off >>= 1) s2 += __shfl_xor(s2, off, 32);
  if ((c & 31) == 0) red[c >> 5] = s2;
  __syncthreads();
  float var = 0.0f;
  #pragma unroll
  for (int i = 0; i < 8; ++i) var += red[i];
  var *= (1.0f / (float)HID);
  float outv = d * rsqrtf(var + 1e-5f) * lng[c] + lnb[c];
  outv = (outv > 0.0f) ? outv : 0.1f * outv;
  g[base + c] += outv;
}

// xc = concat(x_mlp, g)
__global__ void k_concat(const float* __restrict__ a, const float* __restrict__ b,
                         float* __restrict__ xc, int N) {
  int i = blockIdx.x * blockDim.x + threadIdx.x;
  if (i >= N * 2 * HID) return;
  int n = i >> 9;          // /512
  int c = i & 511;
  xc[i] = (c < HID) ? a[(size_t)n * HID + c] : b[(size_t)n * HID + (c - HID)];
}

// final 64 -> 1 dot
__global__ void k_head_final(const float* __restrict__ h, const float* __restrict__ w,
                             const float* __restrict__ b, float* __restrict__ out, int N) {
  int n = blockIdx.x * blockDim.x + threadIdx.x;
  if (n >= N) return;
  float acc = b[0];
  const float* ph = h + (size_t)n * 64;
  #pragma unroll
  for (int i = 0; i < 64; ++i) acc += ph[i] * w[i];
  out[n] = acc;
}

// ------------------------------------------------------------------
// Host orchestration
// ------------------------------------------------------------------
extern "C" void kernel_launch(void* const* d_in, const int* in_sizes, int n_in,
                              void* d_out, int out_size, void* d_ws, size_t ws_size,
                              hipStream_t stream) {
  const float* x       = (const float*)d_in[0];
  const int*   ei      = (const int*)  d_in[1];
  const float* mlp_w1  = (const float*)d_in[2];
  const float* mlp_b1  = (const float*)d_in[3];
  const float* bn1_g   = (const float*)d_in[4];
  const float* bn1_b   = (const float*)d_in[5];
  const float* mlp_w2  = (const float*)d_in[6];
  const float* mlp_b2  = (const float*)d_in[7];
  const float* bn2_g   = (const float*)d_in[8];
  const float* bn2_b   = (const float*)d_in[9];
  const float* mlp_w3  = (const float*)d_in[10];
  const float* mlp_b3  = (const float*)d_in[11];
  const float* gat_wl  = (const float*)d_in[12];
  const float* gat_bl  = (const float*)d_in[13];
  const float* gat_wr  = (const float*)d_in[14];
  const float* gat_br  = (const float*)d_in[15];
  const float* gat_att = (const float*)d_in[16];
  const float* gat_bias= (const float*)d_in[17];
  const float* ln_g    = (const float*)d_in[18];
  const float* ln_b    = (const float*)d_in[19];
  const float* head_w1 = (const float*)d_in[20];
  const float* head_b1 = (const float*)d_in[21];
  const float* hbn1_g  = (const float*)d_in[22];
  const float* hbn1_b  = (const float*)d_in[23];
  const float* head_w2 = (const float*)d_in[24];
  const float* head_b2 = (const float*)d_in[25];
  const float* hbn2_g  = (const float*)d_in[26];
  const float* hbn2_b  = (const float*)d_in[27];
  const float* head_w3 = (const float*)d_in[28];
  const float* head_b3 = (const float*)d_in[29];
  const float* head_w4 = (const float*)d_in[30];
  const float* head_b4 = (const float*)d_in[31];
  float* out = (float*)d_out;

  const int N  = in_sizes[0] / IN_CH;
  const int E  = in_sizes[1] / 2;
  const int E2 = E + N;

  // ---- bump allocator over d_ws ----
  char* ws = (char*)d_ws;
  size_t off = 0;
  auto alloc = [&](size_t bytes) -> char* {
    off = (off + 255) & ~(size_t)255;
    char* p = ws + off;
    off += bytes;
    return p;
  };

  _Float16* wt_mlp1 = (_Float16*)alloc((size_t)HID * IN_CH * 2);
  _Float16* wt_mlp2 = (_Float16*)alloc((size_t)HID * HID * 2);
  _Float16* wt_mlp3 = (_Float16*)alloc((size_t)HID * HID * 2);
  _Float16* wt_gl   = (_Float16*)alloc((size_t)NLAYERS * HID * HID * 2);
  _Float16* wt_gr   = (_Float16*)alloc((size_t)NLAYERS * HID * HID * 2);
  _Float16* wt_h1   = (_Float16*)alloc((size_t)HID * (2 * HID) * 2);
  _Float16* wt_h2   = (_Float16*)alloc((size_t)(HID / 2) * HID * 2);
  _Float16* wt_h3   = (_Float16*)alloc((size_t)(HID / 4) * (HID / 2) * 2);
  float* ssPool     = (float*)alloc(14 * 1024 * sizeof(float));  // 14 slots x (512 scale + 512 shift)

  float* bufA    = (float*)alloc((size_t)N * HID * 4);       // h1 / xl / head h1
  float* bufB    = (float*)alloc((size_t)N * HID * 4);       // h2 / xr / head h2
  float* bufXMLP = (float*)alloc((size_t)N * HID * 4);
  float* bufG    = (float*)alloc((size_t)N * HID * 4);
  float* bufO    = (float*)alloc((size_t)N * HID * 4);       // gat o / head h3
  float* bufXC   = (float*)alloc((size_t)N * 2 * HID * 4);
  float* bufLog  = (float*)alloc((size_t)E2 * HEADS * 4);
  float* bufM    = (float*)alloc((size_t)N * HEADS * 4);
  float* bufS    = (float*)alloc((size_t)N * HEADS * 4);

  auto tr = [&](const float* W, _Float16* Wt, int K, int Nc) {
    int tot = K * Nc;
    k_transpose_f16<<<dim3((tot + 255) / 256), dim3(256), 0, stream>>>(W, Wt, K, Nc);
  };
  auto mk_ss = [&](const float* bias, const float* g, const float* b, int hasbn, int n, int slot) {
    k_make_ss<<<dim3((n + 255) / 256), dim3(256), 0, stream>>>(
        bias, g, b, ssPool + slot * 1024, ssPool + slot * 1024 + 512, n, hasbn);
  };
  auto gemm = [&](const float* A, const _Float16* Wt, float* C, int slot,
                  int M, int K, int Nc, int act) {
    dim3 grid((M + BM - 1) / BM, Nc / BN);
    k_gemm_wmma<<<grid, dim3(256), 0, stream>>>(
        A, Wt, C, ssPool + slot * 1024, ssPool + slot * 1024 + 512, M, K, Nc, act);
  };

  // ---- weight prep (deterministic, re-run every call) ----
  tr(mlp_w1, wt_mlp1, IN_CH, HID);
  tr(mlp_w2, wt_mlp2, HID, HID);
  tr(mlp_w3, wt_mlp3, HID, HID);
  for (int i = 0; i < NLAYERS; ++i) {
    tr(gat_wl + (size_t)i * HID * HID, wt_gl + (size_t)i * HID * HID, HID, HID);
    tr(gat_wr + (size_t)i * HID * HID, wt_gr + (size_t)i * HID * HID, HID, HID);
  }
  tr(head_w1, wt_h1, 2 * HID, HID);
  tr(head_w2, wt_h2, HID, HID / 2);
  tr(head_w3, wt_h3, HID / 2, HID / 4);

  mk_ss(mlp_b1, bn1_g, bn1_b, 1, HID, 0);
  mk_ss(mlp_b2, bn2_g, bn2_b, 1, HID, 1);
  mk_ss(mlp_b3, mlp_b3, mlp_b3, 0, HID, 2);
  for (int i = 0; i < NLAYERS; ++i) {
    mk_ss(gat_bl + i * HID, gat_bl, gat_bl, 0, HID, 3 + i);
    mk_ss(gat_br + i * HID, gat_br, gat_br, 0, HID, 7 + i);
  }
  mk_ss(head_b1, hbn1_g, hbn1_b, 1, HID, 11);
  mk_ss(head_b2, hbn2_g, hbn2_b, 1, HID / 2, 12);
  mk_ss(head_b3, head_b3, head_b3, 0, HID / 4, 13);

  // ---- FeatureMLP ----
  gemm(x,    wt_mlp1, bufA,    0, N, IN_CH, HID, 1);
  gemm(bufA, wt_mlp2, bufB,    1, N, HID,   HID, 1);
  gemm(bufB, wt_mlp3, bufXMLP, 2, N, HID,   HID, 0);
  hipMemcpyAsync(bufG, bufXMLP, (size_t)N * HID * 4, hipMemcpyDeviceToDevice, stream);

  // ---- GATv2 layers ----
  const int ethreads = E2 * HEADS;
  for (int i = 0; i < NLAYERS; ++i) {
    gemm(bufG, wt_gl + (size_t)i * HID * HID, bufA, 3 + i, N, HID, HID, 0); // xl
    gemm(bufG, wt_gr + (size_t)i * HID * HID, bufB, 7 + i, N, HID, HID, 0); // xr
    k_gat_init<<<dim3((N * HID + 255) / 256), dim3(256), 0, stream>>>(bufM, bufS, bufO, N);
    k_edge_logits<<<dim3((ethreads + 255) / 256), dim3(256), 0, stream>>>(
        ei, E, N, bufA, bufB, gat_att + i * HEADS * CDIM, bufLog, bufM);
    k_edge_exp<<<dim3((ethreads + 255) / 256), dim3(256), 0, stream>>>(
        ei, E, N, bufLog, bufM, bufS);
    k_edge_aggr<<<dim3((ethreads + 255) / 256), dim3(256), 0, stream>>>(
        ei, E, N, bufLog, bufS, bufA, bufO);
    k_ln_res<<<dim3(N), dim3(256), 0, stream>>>(
        bufO, gat_bias + i * HID, ln_g + i * HID, ln_b + i * HID, bufG);
  }

  // ---- regression head ----
  k_concat<<<dim3((N * 2 * HID + 255) / 256), dim3(256), 0, stream>>>(bufXMLP, bufG, bufXC, N);
  gemm(bufXC, wt_h1, bufA, 11, N, 2 * HID, HID,     1);
  gemm(bufA,  wt_h2, bufB, 12, N, HID,     HID / 2, 1);
  gemm(bufB,  wt_h3, bufO, 13, N, HID / 2, HID / 4, 1);
  k_head_final<<<dim3((N + 255) / 256), dim3(256), 0, stream>>>(bufO, head_w4, head_b4, out, N);
}
